// HGRN_30236569764164
// MI455X (gfx1250) — compile-verified, matrix operand order checked by reading, not run on hardware
//
#include <hip/hip_runtime.h>

// ---------------------------------------------------------------------------
// HGRN forward for MI455X (gfx1250, wave32, WMMA).
// All GEMMs use v_wmma_f32_16x16x32_bf16 (bf16 A/B, fp32 accumulate).
// Global->LDS staging: CDNA5 async-to-LDS ops, double-buffered; the main loop
// stages tile k+1 then s_wait_asynccnt(4) so DMA overlaps this tile's WMMAs.
// Last K-step is peeled to keep the hot loop branch-free.
// ---------------------------------------------------------------------------

typedef unsigned short u16;
typedef __bf16 bf16_t;
typedef __attribute__((ext_vector_type(16))) bf16_t v16bf;
typedef __attribute__((ext_vector_type(8)))  float  v8f;
typedef __attribute__((ext_vector_type(4)))  int    v4i;   // native vector, not HIP_vector_type

#define DIM_   1024
#define BATCH_ 2
#define SEQ_   2048
#define MTOT_  (BATCH_ * SEQ_)   // 4096 rows
#define VOCAB_ 32000
#define NLAY_  4

#define GLOBAL_AS __attribute__((address_space(1)))
#define LDS_AS    __attribute__((address_space(3)))

typedef GLOBAL_AS v4i* gq_ptr;   // global-AS native int4*
typedef LDS_AS    v4i* lq_ptr;   // LDS-AS native int4*

#if __has_builtin(__builtin_amdgcn_global_load_async_to_lds_b128) && \
    __has_builtin(__builtin_amdgcn_s_wait_asynccnt)
#define USE_ASYNC_LDS 1
#else
#define USE_ASYNC_LDS 0
#endif

// float -> bf16 round-to-nearest-even (bit-level)
static __device__ __forceinline__ u16 f2bf(float f) {
  unsigned int u = __float_as_uint(f);
  unsigned int rnd = ((u >> 16) & 1u) + 0x7FFFu;
  return (u16)((u + rnd) >> 16);
}

// ---------------------------------------------------------------------------
// Tiled WMMA GEMM:  out[M,N] = act( A[M,K] (bf16) * BT[N,K]^T (bf16) + bias )
// Workgroup tile 128x128, 8 waves in 2x4 grid, each wave 64x32 (4x2 WMMA tiles)
// K stepped by 32.  LDS rows padded to 40 halves (80B) to spread banks.
// mode: 0 = fp32 out (nontemporal; head), 1 = silu->fp32,
//       2 = sigmoid(x+bias)->fp32, 3 = fp32 out + bf16 out (dual write)
// ---------------------------------------------------------------------------
__global__ __launch_bounds__(256)
void hgrn_gemm_bf16(const u16* __restrict__ A, const u16* __restrict__ BT,
                    const float* __restrict__ bias,
                    float* __restrict__ outF, u16* __restrict__ outBF,
                    int Mdim, int Ndim, int Kdim, int mode) {
  __shared__ u16 As[2][128 * 40];
  __shared__ u16 Bs[2][128 * 40];

  const int tid  = threadIdx.x;
  const int lane = tid & 31;
  const int wave = tid >> 5;
  const int wmBase = (wave >> 2) * 64;   // 0 or 64
  const int wnBase = (wave & 3) * 32;    // 0,32,64,96
  const int hh = lane >> 4;              // half-wave select
  const int ln = lane & 15;

  const int m0 = blockIdx.y * 128;
  const int n0 = blockIdx.x * 128;

  // staging map: thread covers rows (tid>>2) and (tid>>2)+64, K-chunk (tid&3)*8
  const int srow = tid >> 2;
  const int skc  = (tid & 3) << 3;

  const v8f vzero = {0.f, 0.f, 0.f, 0.f, 0.f, 0.f, 0.f, 0.f};
  v8f acc[4][2];
#pragma unroll
  for (int fm = 0; fm < 4; ++fm)
#pragma unroll
    for (int fn = 0; fn < 2; ++fn) acc[fm][fn] = vzero;

  union Frag { uint4 q[2]; v16bf v; };

  // issue the 4 async (or sync-fallback) copies of one 128x32 tile pair
  auto stage = [&](int k0, int buf) {
    const u16* gA0 = A  + (size_t)(m0 + srow)      * Kdim + k0 + skc;
    const u16* gA1 = A  + (size_t)(m0 + srow + 64) * Kdim + k0 + skc;
    const u16* gB0 = BT + (size_t)(n0 + srow)      * Kdim + k0 + skc;
    const u16* gB1 = BT + (size_t)(n0 + srow + 64) * Kdim + k0 + skc;
    u16* lA0 = &As[buf][srow * 40 + skc];
    u16* lA1 = &As[buf][(srow + 64) * 40 + skc];
    u16* lB0 = &Bs[buf][srow * 40 + skc];
    u16* lB1 = &Bs[buf][(srow + 64) * 40 + skc];
#if USE_ASYNC_LDS
    __builtin_amdgcn_global_load_async_to_lds_b128((gq_ptr)gA0, (lq_ptr)lA0, 0, 0);
    __builtin_amdgcn_global_load_async_to_lds_b128((gq_ptr)gA1, (lq_ptr)lA1, 0, 0);
    __builtin_amdgcn_global_load_async_to_lds_b128((gq_ptr)gB0, (lq_ptr)lB0, 0, 0);
    __builtin_amdgcn_global_load_async_to_lds_b128((gq_ptr)gB1, (lq_ptr)lB1, 0, 0);
#else
    uint4 av0 = *(const uint4*)gA0;
    uint4 av1 = *(const uint4*)gA1;
    uint4 bv0 = *(const uint4*)gB0;
    uint4 bv1 = *(const uint4*)gB1;
    *(uint4*)lA0 = av0;
    *(uint4*)lA1 = av1;
    *(uint4*)lB0 = bv0;
    *(uint4*)lB1 = bv1;
#endif
  };

  // 8 WMMAs on the fragments of LDS buffer `buf`
  // fragments per ISA 16-bit A layout:
  //   lanes 0-15:  v0-3 hold K 0..7,  v4-7 hold K 16..23
  //   lanes 16-31: v0-3 hold K 8..15, v4-7 hold K 24..31
  auto compute = [&](int buf) {
    Frag af[4], bfr[2];
#pragma unroll
    for (int fm = 0; fm < 4; ++fm) {
      int row = wmBase + fm * 16 + ln;
      af[fm].q[0] = *(const uint4*)(&As[buf][row * 40 + 8 * hh]);
      af[fm].q[1] = *(const uint4*)(&As[buf][row * 40 + 16 + 8 * hh]);
    }
#pragma unroll
    for (int fn = 0; fn < 2; ++fn) {
      int row = wnBase + fn * 16 + ln;
      bfr[fn].q[0] = *(const uint4*)(&Bs[buf][row * 40 + 8 * hh]);
      bfr[fn].q[1] = *(const uint4*)(&Bs[buf][row * 40 + 16 + 8 * hh]);
    }
#pragma unroll
    for (int fm = 0; fm < 4; ++fm)
#pragma unroll
      for (int fn = 0; fn < 2; ++fn)
        acc[fm][fn] = __builtin_amdgcn_wmma_f32_16x16x32_bf16(
            false, af[fm].v, false, bfr[fn].v, (short)0, acc[fm][fn],
            false, false);
  };

  int cur = 0;
  stage(0, 0);                            // prologue: DMA first tile into buf 0

  // main loop: branch-free, always stages next tile then waits for current
  for (int k0 = 0; k0 < Kdim - 32; k0 += 32) {
    stage(k0 + 32, cur ^ 1);              // DMA next tile into other buffer
#if USE_ASYNC_LDS
    __builtin_amdgcn_s_wait_asynccnt(4);  // in-order: tile `cur` has landed
#endif
    __syncthreads();                      // tile `cur` visible to all waves
    compute(cur);
    __syncthreads();                      // all waves done reading `cur`
    cur ^= 1;
  }

  // peeled last K-step: wait for everything, no trailing barrier needed
#if USE_ASYNC_LDS
  __builtin_amdgcn_s_wait_asynccnt(0);
#endif
  __syncthreads();
  compute(cur);

  // ---- epilogue: C/D layout = VGPR r holds M = r + 8*hh, lanes = N ----
#pragma unroll
  for (int fm = 0; fm < 4; ++fm) {
#pragma unroll
    for (int fn = 0; fn < 2; ++fn) {
      int gc = n0 + wnBase + fn * 16 + ln;
      float bv = (mode == 2) ? bias[gc] : 0.f;
#pragma unroll
      for (int r = 0; r < 8; ++r) {
        int gr = m0 + wmBase + fm * 16 + r + 8 * hh;
        float v = acc[fm][fn][r];
        if (mode == 1) {                       // silu
          v = v / (1.f + __expf(-v));
        } else if (mode == 2) {                // sigmoid(x + bias)
          v = 1.f / (1.f + __expf(-(v + bv)));
        }
        size_t oix = (size_t)gr * Ndim + gc;
        if (mode == 0) {
          // head logits: 524MB write-once stream -> keep it out of L2
          __builtin_nontemporal_store(v, &outF[oix]);
        } else {
          outF[oix] = v;
          if (mode == 3) outBF[oix] = f2bf(v);
        }
      }
    }
  }
}

// ---------------------------------------------------------------------------
// Weight prep: transpose 1024x1024 fp32 [d][e] -> bf16 [e][d], per layer (z)
// ---------------------------------------------------------------------------
__global__ __launch_bounds__(256)
void hgrn_transpose_w(const float* __restrict__ src, u16* __restrict__ dst) {
  __shared__ float tile[32][33];
  const int l = blockIdx.z;
  const float* s = src + (size_t)l * DIM_ * DIM_;
  u16* d = dst + (size_t)l * 4 * DIM_ * DIM_;   // dst layout [l][mat][e][dd]
  int e0 = blockIdx.x * 32;
  int d0 = blockIdx.y * 32;
#pragma unroll
  for (int j = 0; j < 4; ++j) {
    int dd = d0 + threadIdx.y + j * 8;
    tile[threadIdx.y + j * 8][threadIdx.x] = s[(size_t)dd * DIM_ + e0 + threadIdx.x];
  }
  __syncthreads();
#pragma unroll
  for (int j = 0; j < 4; ++j) {
    int ee = e0 + threadIdx.y + j * 8;
    d[(size_t)ee * DIM_ + d0 + threadIdx.x] = f2bf(tile[threadIdx.x][threadIdx.y + j * 8]);
  }
}

// fp32 -> bf16 straight convert (for Whead, already [N][K] layout)
__global__ void hgrn_convert_bf16(const float* __restrict__ src,
                                  u16* __restrict__ dst, size_t n) {
  size_t i = (size_t)blockIdx.x * blockDim.x + threadIdx.x;
  if (i < n) dst[i] = f2bf(src[i]);
}

// embedding gather -> bf16 activations
__global__ void hgrn_embed(const int* __restrict__ ids,
                           const float* __restrict__ emb,
                           u16* __restrict__ xbf) {
  size_t i = (size_t)blockIdx.x * blockDim.x + threadIdx.x;  // < 4096*1024
  int row = (int)(i >> 10);
  int col = (int)(i & 1023);
  int tok = ids[row];
  xbf[i] = f2bf(emb[(size_t)tok * DIM_ + col]);
}

// gated recurrence h[t] = f[t]*h[t-1] + i[t]*c[t] ; writes h (fp32, in-place
// over cand) and bf16 copy for the Wo GEMM.  One thread per (b, d) chain.
__global__ void hgrn_recurrence(const float* __restrict__ F,
                                const float* __restrict__ I,
                                float* __restrict__ C,
                                u16* __restrict__ hbf) {
  int idx = blockIdx.x * blockDim.x + threadIdx.x;  // < BATCH_*DIM_
  int b = idx / DIM_;
  int dd = idx % DIM_;
  size_t base = (size_t)b * SEQ_ * DIM_ + dd;
  float h = 0.f;
  for (int t = 0; t < SEQ_; ++t) {
    size_t ix = base + (size_t)t * DIM_;
    h = F[ix] * h + I[ix] * C[ix];
    C[ix] = h;
    hbf[ix] = f2bf(h);
  }
}

// LayerNorm over last dim (1024), one block per row, output bf16
__global__ __launch_bounds__(256)
void hgrn_layernorm(const float* __restrict__ X, const float* __restrict__ gamma,
                    const float* __restrict__ beta, u16* __restrict__ out) {
  __shared__ float red[256];
  const int row = blockIdx.x;
  const int tid = threadIdx.x;
  float xv[4];
  float s = 0.f;
#pragma unroll
  for (int j = 0; j < 4; ++j) {
    xv[j] = X[(size_t)row * DIM_ + tid + j * 256];
    s += xv[j];
  }
  red[tid] = s;
  __syncthreads();
  for (int st = 128; st > 0; st >>= 1) {
    if (tid < st) red[tid] += red[tid + st];
    __syncthreads();
  }
  float mean = red[0] * (1.f / DIM_);
  __syncthreads();
  float sq = 0.f;
#pragma unroll
  for (int j = 0; j < 4; ++j) {
    float d = xv[j] - mean;
    sq += d * d;
  }
  red[tid] = sq;
  __syncthreads();
  for (int st = 128; st > 0; st >>= 1) {
    if (tid < st) red[tid] += red[tid + st];
    __syncthreads();
  }
  float rs = rsqrtf(red[0] * (1.f / DIM_) + 1e-5f);
#pragma unroll
  for (int j = 0; j < 4; ++j) {
    int c = tid + j * 256;
    float v = (xv[j] - mean) * rs * gamma[c] + beta[c];
    out[(size_t)row * DIM_ + c] = f2bf(v);
  }
}

// tuple tail: the reference returns (logits, 0.0f)
__global__ void hgrn_write_scalar(float* out, size_t idx) {
  if (threadIdx.x == 0 && blockIdx.x == 0) out[idx] = 0.0f;
}

// ---------------------------------------------------------------------------
extern "C" void kernel_launch(void* const* d_in, const int* in_sizes, int n_in,
                              void* d_out, int out_size, void* d_ws, size_t ws_size,
                              hipStream_t stream) {
  const int*   ids   = (const int*)  d_in[0];   // [2,2048]
  const float* emb   = (const float*)d_in[1];   // [32000,1024]
  const float* Wi    = (const float*)d_in[2];   // [4,1024,1024]
  const float* Wf    = (const float*)d_in[3];
  const float* bf_   = (const float*)d_in[4];   // [4,1024]
  const float* Wg    = (const float*)d_in[5];
  const float* bg    = (const float*)d_in[6];
  const float* Wo    = (const float*)d_in[7];
  const float* gamma = (const float*)d_in[8];
  const float* beta  = (const float*)d_in[9];
  const float* Whead = (const float*)d_in[10];  // [32000,1024]
  float* out = (float*)d_out;

  // ---- workspace layout ----
  char* ws = (char*)d_ws;
  size_t off = 0;
  u16* WT      = (u16*)(ws + off); off += (size_t)NLAY_ * 4 * DIM_ * DIM_ * 2; // 33.6MB
  u16* WheadBF = (u16*)(ws + off); off += (size_t)VOCAB_ * DIM_ * 2;           // 65.5MB
  u16* xbf     = (u16*)(ws + off); off += (size_t)MTOT_ * DIM_ * 2;
  u16* hbf     = (u16*)(ws + off); off += (size_t)MTOT_ * DIM_ * 2;
  u16* normbf  = (u16*)(ws + off); off += (size_t)MTOT_ * DIM_ * 2;
  float* cand  = (float*)(ws + off); off += (size_t)MTOT_ * DIM_ * 4;
  float* fgate = (float*)(ws + off); off += (size_t)MTOT_ * DIM_ * 4;
  float* igate = (float*)(ws + off); off += (size_t)MTOT_ * DIM_ * 4;
  float* xf    = (float*)(ws + off); off += (size_t)MTOT_ * DIM_ * 4;
  (void)ws_size;

  const size_t matElems = (size_t)DIM_ * DIM_;

  // ---- weight prep: transpose+convert per-layer mats; convert head ----
  dim3 tb(32, 8), tg(DIM_ / 32, DIM_ / 32, NLAY_);
  hgrn_transpose_w<<<tg, tb, 0, stream>>>(Wi, WT + 0 * matElems);
  hgrn_transpose_w<<<tg, tb, 0, stream>>>(Wf, WT + 1 * matElems);
  hgrn_transpose_w<<<tg, tb, 0, stream>>>(Wg, WT + 2 * matElems);
  hgrn_transpose_w<<<tg, tb, 0, stream>>>(Wo, WT + 3 * matElems);

  size_t headN = (size_t)VOCAB_ * DIM_;
  hgrn_convert_bf16<<<(unsigned)((headN + 255) / 256), 256, 0, stream>>>(
      Whead, WheadBF, headN);

  hgrn_embed<<<(MTOT_ * DIM_) / 256, 256, 0, stream>>>(ids, emb, xbf);

  // ---- layers ----
  dim3 gLayer(DIM_ / 128, MTOT_ / 128);   // 8 x 32
  for (int l = 0; l < NLAY_; ++l) {
    const u16* wt = WT + (size_t)l * 4 * matElems;
    hgrn_gemm_bf16<<<gLayer, 256, 0, stream>>>(xbf, wt + 0 * matElems, nullptr,
                                               cand, nullptr, MTOT_, DIM_, DIM_, 1);
    hgrn_gemm_bf16<<<gLayer, 256, 0, stream>>>(xbf, wt + 1 * matElems, bf_ + l * DIM_,
                                               fgate, nullptr, MTOT_, DIM_, DIM_, 2);
    hgrn_gemm_bf16<<<gLayer, 256, 0, stream>>>(xbf, wt + 2 * matElems, bg + l * DIM_,
                                               igate, nullptr, MTOT_, DIM_, DIM_, 2);
    hgrn_recurrence<<<(BATCH_ * DIM_) / 256, 256, 0, stream>>>(fgate, igate, cand, hbf);
    hgrn_gemm_bf16<<<gLayer, 256, 0, stream>>>(hbf, wt + 3 * matElems, nullptr,
                                               xf, xbf, MTOT_, DIM_, DIM_, 3);
  }

  // ---- layernorm + vocab head ----
  hgrn_layernorm<<<MTOT_, 256, 0, stream>>>(xf, gamma, beta, normbf);

  dim3 gHead(VOCAB_ / 128, MTOT_ / 128);  // 250 x 32
  hgrn_gemm_bf16<<<gHead, 256, 0, stream>>>(normbf, WheadBF, nullptr,
                                            out, nullptr, MTOT_, VOCAB_, DIM_, 0);

  hgrn_write_scalar<<<1, 1, 0, stream>>>(out, (size_t)MTOT_ * VOCAB_);
}
